// MyModel_49417893708135
// MI455X (gfx1250) — matrix operand order repaired and checked by reading, not compiled, not run on hardware
//
#include <hip/hip_runtime.h>
#include <hip/hip_bf16.h>
#include <stdint.h>

// ---------------------------------------------------------------------------
// MI455X (gfx1250) wave32 WMMA attention pipeline, bf16 compute / f32 accum.
// ---------------------------------------------------------------------------

typedef __attribute__((ext_vector_type(16))) __bf16 v16bf;
typedef __attribute__((ext_vector_type(8)))  float  v8f;

struct Frag16 { uint4 lo, hi; };            // 16 bf16 elements (32 bytes)
static_assert(sizeof(Frag16) == 32, "frag size");

__device__ __forceinline__ v16bf as_v16bf(const Frag16& f) {
    return __builtin_bit_cast(v16bf, f);
}

__device__ __forceinline__ unsigned short f32_to_bf16(float f) {
    unsigned u = __builtin_bit_cast(unsigned, f);
    u += 0x7FFFu + ((u >> 16) & 1u);        // round-to-nearest-even
    return (unsigned short)(u >> 16);
}

__device__ __forceinline__ void wait_asynccnt0() {
#if __has_builtin(__builtin_amdgcn_s_wait_asynccnt)
    __builtin_amdgcn_s_wait_asynccnt(0);
#else
    asm volatile("s_wait_asynccnt 0x0" ::: "memory");
#endif
}

// global -> LDS async DMA, 16 bytes, no VGPR bounce (ASYNCcnt tracked)
__device__ __forceinline__ void async_copy_b128(unsigned lds_off, const void* g) {
    asm volatile("global_load_async_to_lds_b128 %0, %1, off"
                 :: "v"(lds_off), "v"((unsigned long long)(uintptr_t)g)
                 : "memory");
}

// ---------------------------------------------------------------------------
// Elementwise f32 -> bf16 convert (streaming: NT loads)
// ---------------------------------------------------------------------------
__global__ void cvt_bf16_kernel(const float* __restrict__ in,
                                unsigned short* __restrict__ out, int n) {
    int i = blockIdx.x * 256 + threadIdx.x;
    if (i < n) out[i] = f32_to_bf16(__builtin_nontemporal_load(in + i));
}

// Transpose + convert: in [K][N] f32 -> out [N][K] bf16
__global__ void tr_bf16_kernel(const float* __restrict__ in,
                               unsigned short* __restrict__ out, int K, int N) {
    int id = blockIdx.x * 256 + threadIdx.x;
    if (id >= K * N) return;
    int n = id / K, k = id % K;
    out[id] = f32_to_bf16(in[(size_t)k * N + n]);
}

// ---------------------------------------------------------------------------
// bf16 WMMA GEMM, 32x32 output tile per wave (2x2 register blocking:
// 2 A-frags x 2 B-frags -> 4 WMMAs per 32-deep K-chunk).  B given transposed
// as BT[N][K].  4 waves per block.
// mode 0: Q-projection  -> (acc+b)*scale, scatter to Qb [B,NH,Lq,HD] bf16
// mode 1: KV-projection -> scatter to Kb / Vb [B,NH,Lkv,HD] bf16
// mode 2: out-projection-> acc+b, f32 row-major output
// ---------------------------------------------------------------------------
__global__ __launch_bounds__(128) void gemm32_kernel(
        const unsigned short* __restrict__ A,
        const unsigned short* __restrict__ BT,
        const float* __restrict__ bias,
        void* __restrict__ outp, void* __restrict__ outp2,
        int M, int N, int K, int mode)
{
    const int lane = threadIdx.x & 31;
    const int wave = threadIdx.x >> 5;
    const int half = lane >> 4;
    const int l16  = lane & 15;

    const int ntx  = N >> 5;                 // 32-wide column tiles
    const int tile = blockIdx.x * 4 + wave;
    if (tile >= (M >> 5) * ntx) return;      // wave-uniform: EXEC stays all-1
    const int r0 = (tile / ntx) * 32;
    const int n0 = (tile % ntx) * 32;

    const unsigned short* ar[2] = { A  + (size_t)(r0 + l16) * K,
                                    A  + (size_t)(r0 + 16 + l16) * K };
    const unsigned short* br[2] = { BT + (size_t)(n0 + l16) * K,
                                    BT + (size_t)(n0 + 16 + l16) * K };

    v8f acc[2][2];
#pragma unroll
    for (int ti = 0; ti < 2; ti++)
#pragma unroll
        for (int tj = 0; tj < 2; tj++) acc[ti][tj] = v8f{};

#pragma unroll 2
    for (int k0 = 0; k0 < K; k0 += 32) {
        Frag16 af[2], bf[2];
#pragma unroll
        for (int t = 0; t < 2; t++) {
            // A 16x32: lanes 0-15 hold K {0..7,16..23}; lanes 16-31 {8..15,24..31}
            af[t].lo = *(const uint4*)(ar[t] + k0 + half * 8);
            af[t].hi = *(const uint4*)(ar[t] + k0 + 16 + half * 8);
            // B 32x16: lanes 0-15 hold K {0..7,16..23}; lanes 16-31 {8..15,24..31}
            bf[t].lo = *(const uint4*)(br[t] + k0 + half * 8);
            bf[t].hi = *(const uint4*)(br[t] + k0 + 16 + half * 8);
        }
#pragma unroll
        for (int ti = 0; ti < 2; ti++)
#pragma unroll
            for (int tj = 0; tj < 2; tj++)
                acc[ti][tj] = __builtin_amdgcn_wmma_f32_16x16x32_bf16(
                                  false, as_v16bf(af[ti]), false, as_v16bf(bf[tj]),
                                  (short)0, acc[ti][tj], false, false);
    }

    if (mode == 0) {
        unsigned short* Qb = (unsigned short*)outp;
#pragma unroll
        for (int tj = 0; tj < 2; tj++) {
            int e  = n0 + tj * 16 + l16;
            float bv = bias[e];
            int h = e >> 6, hd = e & 63;
#pragma unroll
            for (int ti = 0; ti < 2; ti++)
#pragma unroll
                for (int i = 0; i < 8; i++) {
                    int r = r0 + ti * 16 + half * 8 + i;
                    int b = r >> 10, q = r & 1023;
                    Qb[(((size_t)b * 8 + h) * 1024 + q) * 64 + hd] =
                        f32_to_bf16((acc[ti][tj][i] + bv) * 0.125f);  // HD^-0.5
                }
        }
    } else if (mode == 1) {
        unsigned short* Kb = (unsigned short*)outp;
        unsigned short* Vb = (unsigned short*)outp2;
#pragma unroll
        for (int tj = 0; tj < 2; tj++) {
            int c  = n0 + tj * 16 + l16;
            float bv = bias[c];
            int s = c >> 9, e = c & 511;
            int h = e >> 6, hd = e & 63;
            unsigned short* dst = s ? Vb : Kb;
#pragma unroll
            for (int ti = 0; ti < 2; ti++)
#pragma unroll
                for (int i = 0; i < 8; i++) {
                    int r = r0 + ti * 16 + half * 8 + i;
                    int b = r >> 12, l = r & 4095;
                    dst[(((size_t)b * 8 + h) * 4096 + l) * 64 + hd] =
                        f32_to_bf16(acc[ti][tj][i] + bv);
                }
        }
    } else {
        float* o = (float*)outp;
#pragma unroll
        for (int tj = 0; tj < 2; tj++) {
            int c  = n0 + tj * 16 + l16;
            float bv = bias[c];
#pragma unroll
            for (int ti = 0; ti < 2; ti++)
#pragma unroll
                for (int i = 0; i < 8; i++) {
                    int r = r0 + ti * 16 + half * 8 + i;
                    o[(size_t)r * N + c] = acc[ti][tj][i] + bv;
                }
        }
    }
}

// ---------------------------------------------------------------------------
// Flash attention: block = 4 waves x 32 = 128 threads; each wave owns a
// 16-query tile; waves share one (b,h) and cooperatively stage K / V^T tiles
// (32 keys) through LDS.  K tile uses async global->LDS DMA; online softmax
// is seeded with the dummy column.
// ---------------------------------------------------------------------------
#define KSTRIDE 72   // LDS row stride for K tile   (bf16 elems, 144B = 16B-aligned)
#define VSTRIDE 48   // LDS row stride for V^T tile (96B, 16B-aligned)
#define PSTRIDE 48   // LDS row stride for P tile

__global__ __launch_bounds__(128) void attn_kernel(
        const unsigned short* __restrict__ Qb,
        const unsigned short* __restrict__ Kb,
        const unsigned short* __restrict__ Vb,
        const int*   __restrict__ mask,
        const float* __restrict__ pos,       // [NH,Lq,Lkv]
        unsigned short* __restrict__ AO)     // [B,Lq,E] bf16
{
    __shared__ unsigned short ldsK[32 * KSTRIDE];
    __shared__ unsigned short ldsVT[64 * VSTRIDE];
    __shared__ unsigned short ldsP[4][16 * PSTRIDE];

    const int tid  = threadIdx.x;
    const int wave = tid >> 5;
    const int lane = tid & 31;
    const int half = lane >> 4;
    const int l16  = lane & 15;

    const int idx  = blockIdx.x;
    const int qblk = idx & 15;
    const int h    = (idx >> 4) & 7;
    const int b    = idx >> 7;
    const int q0   = qblk * 64 + wave * 16;

    // --- Q fragments (A-layout), loaded once: 16 queries x 64 hd ---
    const unsigned short* Qrow =
        Qb + (((size_t)(b * 8 + h) * 1024) + q0 + l16) * 64;
    Frag16 qa[2];
    qa[0].lo = *(const uint4*)(Qrow + half * 8);
    qa[0].hi = *(const uint4*)(Qrow + 16 + half * 8);
    qa[1].lo = *(const uint4*)(Qrow + 32 + half * 8);
    qa[1].hi = *(const uint4*)(Qrow + 48 + half * 8);

    // --- per-row online-softmax state (rows m = half*8 + i) ---
    float Mrow[8], Srow[8], rb[8];
    v8f O[4];
#pragma unroll
    for (int nc = 0; nc < 4; nc++) O[nc] = v8f{};
#pragma unroll
    for (int i = 0; i < 8; i++) {
        int m  = half * 8 + i;
        int mk = mask[b * 1024 + q0 + m];
        rb[i]   = mk ? 0.0f : -100.0f;       // row bias on real scores
        Mrow[i] = mk ? -100.0f : 100.0f;     // dummy column seeds the max
        Srow[i] = 1.0f;                      // exp(dummy - M0) = 1
    }

    const unsigned short* Kbase = Kb + ((size_t)(b * 8 + h) * 4096) * 64;
    const unsigned short* Vbase = Vb + ((size_t)(b * 8 + h) * 4096) * 64;
    const float* posbase = pos + ((size_t)h * 1024 + q0) * 4096;

    for (int k0 = 0; k0 < 4096; k0 += 32) {
        // ---- stage K (async DMA, no VGPR bounce) and V^T into LDS ----
#pragma unroll
        for (int p = 0; p < 2; p++) {
            int id  = p * 128 + tid;         // 0..255 => 32 rows x 8 segs
            int row = id >> 3, seg = id & 7;

            unsigned kofs = (unsigned)(uintptr_t)&ldsK[row * KSTRIDE + seg * 8];
            async_copy_b128(kofs, Kbase + (size_t)(k0 + row) * 64 + seg * 8);

            uint4 vv = *(const uint4*)(Vbase + (size_t)(k0 + row) * 64 + seg * 8);
            const unsigned short* vs = (const unsigned short*)&vv;
#pragma unroll
            for (int j = 0; j < 8; j++)
                ldsVT[(seg * 8 + j) * VSTRIDE + row] = vs[j];

            if (k0 + 32 < 4096) {            // global_prefetch next tile
                __builtin_prefetch(Kbase + (size_t)(k0 + 32 + row) * 64 + seg * 8, 0, 3);
                __builtin_prefetch(Vbase + (size_t)(k0 + 32 + row) * 64 + seg * 8, 0, 3);
            }
        }
        wait_asynccnt0();                    // drain this wave's async copies
        __syncthreads();

        // ---- S = Q @ K^T : two 16x16 tiles (keys k0..+15, k0+16..+31) ----
        v8f st[2]; st[0] = v8f{}; st[1] = v8f{};
#pragma unroll
        for (int c = 0; c < 2; c++) {        // hd chunks of 32
            v16bf a = as_v16bf(qa[c]);
#pragma unroll
            for (int t = 0; t < 2; t++) {
                Frag16 bf;
                const unsigned short* kr =
                    &ldsK[(t * 16 + l16) * KSTRIDE + c * 32 + half * 8];
                bf.lo = *(const uint4*)(kr);
                bf.hi = *(const uint4*)(kr + 16);
                st[t] = __builtin_amdgcn_wmma_f32_16x16x32_bf16(
                            false, a, false, as_v16bf(bf),
                            (short)0, st[t], false, false);
            }
        }

        // ---- + attn_pos (NT stream), + row bias, online softmax update ----
        float alpha[8];
#pragma unroll
        for (int i = 0; i < 8; i++) {
            int m = half * 8 + i;
            float p0 = st[0][i] + rb[i] +
                __builtin_nontemporal_load(posbase + (size_t)m * 4096 + k0 + l16);
            float p1 = st[1][i] + rb[i] +
                __builtin_nontemporal_load(posbase + (size_t)m * 4096 + k0 + 16 + l16);
            float mx = fmaxf(p0, p1);
#pragma unroll
            for (int off = 8; off > 0; off >>= 1)
                mx = fmaxf(mx, __shfl_xor(mx, off, 32));
            float Mn = fmaxf(Mrow[i], mx);
            float e0 = __expf(p0 - Mn);
            float e1 = __expf(p1 - Mn);
            float rs = e0 + e1;
#pragma unroll
            for (int off = 8; off > 0; off >>= 1)
                rs += __shfl_xor(rs, off, 32);
            float al = __expf(Mrow[i] - Mn);
            Srow[i] = Srow[i] * al + rs;
            Mrow[i] = Mn;
            alpha[i] = al;
            // P tile (C-layout -> LDS, re-read below in A-layout)
            ldsP[wave][m * PSTRIDE + l16]      = f32_to_bf16(e0);
            ldsP[wave][m * PSTRIDE + 16 + l16] = f32_to_bf16(e1);
        }
#pragma unroll
        for (int nc = 0; nc < 4; nc++)
#pragma unroll
            for (int i = 0; i < 8; i++) O[nc][i] *= alpha[i];

        // ---- P fragment (A-layout, 16x32) from wave-private LDS ----
        Frag16 pf;
        const unsigned short* pr = &ldsP[wave][l16 * PSTRIDE + half * 8];
        pf.lo = *(const uint4*)(pr);
        pf.hi = *(const uint4*)(pr + 16);
        v16bf pa = as_v16bf(pf);

        // ---- O += P @ V : four 16-wide hd chunks, K = 32 keys ----
#pragma unroll
        for (int nc = 0; nc < 4; nc++) {
            Frag16 vf;
            const unsigned short* vr =
                &ldsVT[(nc * 16 + l16) * VSTRIDE + half * 8];
            vf.lo = *(const uint4*)(vr);
            vf.hi = *(const uint4*)(vr + 16);
            O[nc] = __builtin_amdgcn_wmma_f32_16x16x32_bf16(
                        false, pa, false, as_v16bf(vf),
                        (short)0, O[nc], false, false);
        }
        __syncthreads();   // before restaging K/V
    }

    // ---- epilogue: O / S, store to [B,Lq,E] (E = h*64 + hd) ----
    unsigned short* outw = AO + ((size_t)b * 1024 + q0) * 512 + h * 64;
#pragma unroll
    for (int i = 0; i < 8; i++) {
        int m = half * 8 + i;
        float inv = 1.0f / Srow[i];
#pragma unroll
        for (int nc = 0; nc < 4; nc++)
            outw[(size_t)m * 512 + nc * 16 + l16] = f32_to_bf16(O[nc][i] * inv);
    }
}

// ---------------------------------------------------------------------------
// Host-side launch
// ---------------------------------------------------------------------------
extern "C" void kernel_launch(void* const* d_in, const int* in_sizes, int n_in,
                              void* d_out, int out_size, void* d_ws, size_t ws_size,
                              hipStream_t stream) {
    (void)in_sizes; (void)n_in; (void)out_size; (void)ws_size;

    const float* Xq   = (const float*)d_in[0];
    const float* Xkv  = (const float*)d_in[1];
    const int*   mask = (const int*)  d_in[2];
    const float* Wq   = (const float*)d_in[3];
    const float* bq   = (const float*)d_in[4];
    const float* Wkv  = (const float*)d_in[5];
    const float* bkv  = (const float*)d_in[6];
    const float* Wp   = (const float*)d_in[7];
    const float* bp   = (const float*)d_in[8];
    const float* pos  = (const float*)d_in[9];
    float* out = (float*)d_out;

    // workspace layout (bf16 element offsets)
    unsigned short* Xq16  = (unsigned short*)d_ws;
    unsigned short* Xkv16 = Xq16  + (size_t)4096 * 512;
    unsigned short* WqT   = Xkv16 + (size_t)16384 * 512;
    unsigned short* WkvT  = WqT   + (size_t)512 * 512;
    unsigned short* WpT   = WkvT  + (size_t)512 * 1024;
    unsigned short* Qb    = WpT   + (size_t)512 * 512;
    unsigned short* Kb    = Qb    + (size_t)4096 * 512;
    unsigned short* Vb    = Kb    + (size_t)16384 * 512;
    unsigned short* AO    = Vb    + (size_t)16384 * 512;

    // precision conversion + weight transposes
    cvt_bf16_kernel<<<(4096 * 512 + 255) / 256, 256, 0, stream>>>(Xq, Xq16, 4096 * 512);
    cvt_bf16_kernel<<<(16384 * 512 + 255) / 256, 256, 0, stream>>>(Xkv, Xkv16, 16384 * 512);
    tr_bf16_kernel<<<(512 * 512 + 255) / 256, 256, 0, stream>>>(Wq,  WqT,  512, 512);
    tr_bf16_kernel<<<(512 * 1024 + 255) / 256, 256, 0, stream>>>(Wkv, WkvT, 512, 1024);
    tr_bf16_kernel<<<(512 * 512 + 255) / 256, 256, 0, stream>>>(Wp,  WpT,  512, 512);

    // Q projection: [4096,512] @ [512,512], fused scale, scatter to heads
    gemm32_kernel<<<512, 128, 0, stream>>>(Xq16, WqT, bq, Qb, nullptr,
                                           4096, 512, 512, 0);
    // KV projection: [16384,512] @ [512,1024], scatter K / V planes
    gemm32_kernel<<<4096, 128, 0, stream>>>(Xkv16, WkvT, bkv, Kb, Vb,
                                            16384, 1024, 512, 1);
    // flash attention: B*NH*(Lq/64) = 512 blocks
    attn_kernel<<<512, 128, 0, stream>>>(Qb, Kb, Vb, mask, pos, AO);
    // output projection: [4096,512] @ [512,512] + bp -> f32
    gemm32_kernel<<<512, 128, 0, stream>>>(AO, WpT, bp, out, nullptr,
                                           4096, 512, 512, 2);
}
